// HNetEnergyViaHamiltonian_42271068127503
// MI455X (gfx1250) — matrix operand order repaired and checked by reading, not compiled, not run on hardware
//
#include <hip/hip_runtime.h>
#include <stdint.h>

// ---------------------------------------------------------------------------
// energies[p,c] = x_p^T H_c x_p + k_c ; out = max(energies) - energies
// P=8192, C=64, N=512.  Compute-bound (275 GFLOP vs 82MB).
// Hot loop: v_wmma_f32_16x16x32_bf16 fed by ds_load_b128 with imm16 offsets
// from asm-pinned base VGPRs + fully unrolled K loop (no VALU addr math,
// no WMMA->VALU hazard NOPs).  H panels double-buffered via
// global_load_async_to_lds_b128 (ASYNCcnt).
// ---------------------------------------------------------------------------

#define N_PTS   8192
#define N_CMP   64
#define N_NODES 512

typedef __bf16 bf16_t;
typedef bf16_t v16bf __attribute__((ext_vector_type(16)));
typedef float  v8f   __attribute__((ext_vector_type(8)));

// ---- workspace layout (bytes) ----
static constexpr size_t XB_OFF    = 0;                                   // bf16 X  [8192][512]
static constexpr size_t XB_BYTES  = (size_t)N_PTS * N_NODES * 2;         // 8 MB
static constexpr size_t HBT_OFF   = XB_OFF + XB_BYTES;                   // bf16 H^T [64][512(m)][512(n)]
static constexpr size_t HBT_BYTES = (size_t)N_CMP * N_NODES * N_NODES*2; // 32 MB
static constexpr size_t EN_OFF    = HBT_OFF + HBT_BYTES;                 // f32 energies [8192][64]
static constexpr size_t EN_BYTES  = (size_t)N_PTS * N_CMP * 4;           // 2 MB
static constexpr size_t BMAX_OFF  = EN_OFF + EN_BYTES;                   // f32 [256]
static constexpr size_t GMAX_OFF  = BMAX_OFF + 256 * 4;                  // f32 [1]

__device__ __forceinline__ uint16_t f2bf(float f) {
  uint32_t u = __float_as_uint(f);
  uint32_t r = u + 0x7FFFu + ((u >> 16) & 1u);   // round-to-nearest-even
  return (uint16_t)(r >> 16);
}
__device__ __forceinline__ float bf2f(uint16_t b) {
  return __uint_as_float(((uint32_t)b) << 16);
}

// ---- CDNA5 async global->LDS copy (16B per lane, ASYNCcnt-tracked) ----
__device__ __forceinline__ void async_cp16(const uint16_t* dst_lds,
                                           const uint16_t* src_global) {
  uint32_t lofs = (uint32_t)(uintptr_t)dst_lds;  // low 32 bits = LDS byte offset
  asm volatile("global_load_async_to_lds_b128 %0, %1, off"
               :
               : "v"(lofs), "v"(src_global)
               : "memory");
}
__device__ __forceinline__ void wait_async0() {
  asm volatile("s_wait_asynccnt 0" ::: "memory");
}

// ---- pass 1: X f32 -> bf16 ----
__global__ void cvt_x_kernel(const float* __restrict__ X,
                             uint16_t* __restrict__ Xb, int n) {
  for (int i = blockIdx.x * blockDim.x + threadIdx.x; i < n;
       i += gridDim.x * blockDim.x)
    Xb[i] = f2bf(X[i]);
}

// ---- pass 2: H[c][n][m] f32 -> Hbt[c][m][n] bf16 (tiled transpose) ----
__global__ void cvt_ht_kernel(const float* __restrict__ H,
                              uint16_t* __restrict__ Hbt) {
  __shared__ float tile[32][33];
  const int c  = blockIdx.z;
  const int m0 = blockIdx.x * 32;
  const int n0 = blockIdx.y * 32;
  const float*  Hc  = H   + (size_t)c * N_NODES * N_NODES;
  uint16_t*     Hct = Hbt + (size_t)c * N_NODES * N_NODES;
  for (int i = threadIdx.y; i < 32; i += 8)
    tile[i][threadIdx.x] = Hc[(size_t)(n0 + i) * N_NODES + m0 + threadIdx.x];
  __syncthreads();
  for (int i = threadIdx.y; i < 32; i += 8)
    Hct[(size_t)(m0 + i) * N_NODES + n0 + threadIdx.x] =
        f2bf(tile[threadIdx.x][i]);
}

// ---- pass 3: fused quad-form via WMMA ----
#define THREADS 128            // 4 waves; each wave owns 32 rows (2 A tiles)
#define PTILE   128            // rows of X per workgroup
#define PANEL   64             // H columns staged per iteration
#define NPANEL  (N_NODES / PANEL)
#define XSTR    (N_NODES + 8)  // padded LDS stride (16B aligned rows)
#define HSTR    (N_NODES + 8)
#define HBASE0  (PTILE * XSTR)                 // u16 offset of H buffer 0
#define HBUFSZ  (PANEL * HSTR)                 // u16 size of one H buffer
#define SMEM_BYTES ((PTILE * XSTR + 2 * PANEL * HSTR) * 2)

__global__ __launch_bounds__(THREADS) void quad_kernel(
    const uint16_t* __restrict__ Xb, const uint16_t* __restrict__ Hbt,
    const float* __restrict__ kvec, float* __restrict__ energies) {
  extern __shared__ uint16_t lds[];   // [0, HBASE0): X tile ; then 2 H buffers

  const int pbase = blockIdx.x * PTILE;
  const int c     = blockIdx.y;
  const int tid   = threadIdx.x;
  const int lane  = tid & 31;
  const int wave  = tid >> 5;
  const int lrow  = lane & 15;                 // M (A) / N (B,C)
  const int lhalf = lane >> 4;
  const int wrow  = wave * 32;                 // 32 rows per wave

  const uint16_t* Hc = Hbt + (size_t)c * N_NODES * N_NODES;

  // prologue: async-stage X tile + H panel 0, then wait + barrier
  for (int i = tid; i < PTILE * 64; i += THREADS) {
    const int r = i >> 6, q = i & 63;
    async_cp16(&lds[r * XSTR + q * 8],
               Xb + (size_t)(pbase + r) * N_NODES + q * 8);
  }
  for (int i = tid; i < PANEL * 64; i += THREADS) {
    const int r = i >> 6, q = i & 63;
    async_cp16(&lds[HBASE0 + r * HSTR + q * 8],
               Hc + (size_t)r * N_NODES + q * 8);
  }
  wait_async0();
  __syncthreads();

  float part0[8], part1[8];
#pragma unroll
  for (int v = 0; v < 8; ++v) { part0[v] = 0.0f; part1[v] = 0.0f; }

  // A base: pinned once; all K/tile deltas (<=17.6KB) fold into ds imm16
  int ab = (wrow + lrow) * XSTR + lhalf * 8;
  asm volatile("" : "+v"(ab));   // opaque -> compiler keeps full value as base

  int buf = 0;
  for (int panel = 0; panel < NPANEL; ++panel) {
    // prefetch next panel into the other buffer (async, overlaps WMMA)
    if (panel + 1 < NPANEL) {
      const uint16_t* src = Hc + (size_t)(panel + 1) * PANEL * N_NODES;
      const int dbase = HBASE0 + (buf ^ 1) * HBUFSZ;
      for (int i = tid; i < PANEL * 64; i += THREADS) {
        const int r = i >> 6, q = i & 63;
        async_cp16(&lds[dbase + r * HSTR + q * 8],
                   src + (size_t)r * N_NODES + q * 8);
      }
    }

    // B base for THIS panel buffer: pinned so HBASE0 stays in the register
    // and mb/kb deltas (<=~51KB) fold into ds imm16 offsets
    int bb = HBASE0 + buf * HBUFSZ + lrow * HSTR + lhalf * 8;
    asm volatile("" : "+v"(bb));

    v8f acc0[4], acc1[4];
#pragma unroll
    for (int mb = 0; mb < 4; ++mb) { acc0[mb] = v8f{}; acc1[mb] = v8f{}; }

    // full K=512, fully unrolled: every ds offset is a literal imm16
#pragma unroll
    for (int kb = 0; kb < N_NODES; kb += 32) {
      union { uint4 u[2]; v16bf v; } A0, A1;
      A0.u[0] = *reinterpret_cast<const uint4*>(&lds[ab + kb]);
      A0.u[1] = *reinterpret_cast<const uint4*>(&lds[ab + kb + 16]);
      A1.u[0] = *reinterpret_cast<const uint4*>(&lds[ab + 16 * XSTR + kb]);
      A1.u[1] = *reinterpret_cast<const uint4*>(&lds[ab + 16 * XSTR + kb + 16]);
#pragma unroll
      for (int mb = 0; mb < 4; ++mb) {
        union { uint4 u[2]; v16bf v; } B;
        B.u[0] = *reinterpret_cast<const uint4*>(&lds[bb + mb * 16 * HSTR + kb]);
        B.u[1] = *reinterpret_cast<const uint4*>(&lds[bb + mb * 16 * HSTR + kb + 16]);
        acc0[mb] = __builtin_amdgcn_wmma_f32_16x16x32_bf16(
            false, A0.v, false, B.v, (short)0, acc0[mb], false, false);
        acc1[mb] = __builtin_amdgcn_wmma_f32_16x16x32_bf16(
            false, A1.v, false, B.v, (short)0, acc1[mb], false, false);
      }
    }

    // elementwise: quad += Z[p,m] * X[p,m]
    // C layout: VGPR v, lane -> M = v + 8*lhalf, N = lane&15
#pragma unroll
    for (int mb = 0; mb < 4; ++mb) {
      const int mcol = panel * PANEL + mb * 16 + lrow;
#pragma unroll
      for (int v = 0; v < 8; ++v) {
        const int row0 = wrow + v + 8 * lhalf;
        part0[v] += acc0[mb][v] * bf2f(lds[row0 * XSTR + mcol]);
        part1[v] += acc1[mb][v] * bf2f(lds[(row0 + 16) * XSTR + mcol]);
      }
    }

    wait_async0();     // our async fills of the other buffer are done
    __syncthreads();   // everyone finished reading buf & filling buf^1
    buf ^= 1;
  }

  // reduce over the 16 N-lanes sharing each row
#pragma unroll
  for (int v = 0; v < 8; ++v) {
    float s0 = part0[v], s1 = part1[v];
#pragma unroll
    for (int off = 8; off > 0; off >>= 1) {
      s0 += __shfl_xor(s0, off, 16);
      s1 += __shfl_xor(s1, off, 16);
    }
    part0[v] = s0; part1[v] = s1;
  }
  if (lrow == 0) {
    const float kc = kvec[c];
#pragma unroll
    for (int v = 0; v < 8; ++v) {
      const int prow = pbase + wrow + v + 8 * lhalf;
      energies[(size_t)prow * N_CMP + c]        = part0[v] + kc;
      energies[(size_t)(prow + 16) * N_CMP + c] = part1[v] + kc;
    }
  }
}

// ---- pass 4/5: two-stage max reduction ----
__global__ void max1_kernel(const float* __restrict__ en,
                            float* __restrict__ bmax, int n) {
  __shared__ float red[256];
  float m = -3.0e38f;
  for (int i = blockIdx.x * 256 + threadIdx.x; i < n; i += gridDim.x * 256)
    m = fmaxf(m, en[i]);
  red[threadIdx.x] = m;
  __syncthreads();
  for (int s = 128; s > 0; s >>= 1) {
    if (threadIdx.x < s)
      red[threadIdx.x] = fmaxf(red[threadIdx.x], red[threadIdx.x + s]);
    __syncthreads();
  }
  if (threadIdx.x == 0) bmax[blockIdx.x] = red[0];
}

__global__ void max2_kernel(const float* __restrict__ bmax,
                            float* __restrict__ gmax, int n) {
  __shared__ float red[256];
  red[threadIdx.x] = (threadIdx.x < n) ? bmax[threadIdx.x] : -3.0e38f;
  __syncthreads();
  for (int s = 128; s > 0; s >>= 1) {
    if (threadIdx.x < s)
      red[threadIdx.x] = fmaxf(red[threadIdx.x], red[threadIdx.x + s]);
    __syncthreads();
  }
  if (threadIdx.x == 0) gmax[0] = red[0];
}

// ---- pass 6: out = gmax - energies ----
__global__ void finalize_kernel(const float* __restrict__ en,
                                const float* __restrict__ gmax,
                                float* __restrict__ out, int n) {
  const float g = gmax[0];
  for (int i = blockIdx.x * blockDim.x + threadIdx.x; i < n;
       i += gridDim.x * blockDim.x)
    out[i] = g - en[i];
}

extern "C" void kernel_launch(void* const* d_in, const int* in_sizes, int n_in,
                              void* d_out, int out_size, void* d_ws,
                              size_t ws_size, hipStream_t stream) {
  (void)in_sizes; (void)n_in; (void)out_size; (void)ws_size;
  const float* X  = (const float*)d_in[0];   // [8192][512]
  const float* H  = (const float*)d_in[1];   // [64][512][512]
  const float* kv = (const float*)d_in[2];   // [64]
  float* out = (float*)d_out;                // [8192][64]

  char* ws = (char*)d_ws;
  uint16_t* Xb   = (uint16_t*)(ws + XB_OFF);
  uint16_t* Hbt  = (uint16_t*)(ws + HBT_OFF);
  float*    en   = (float*)(ws + EN_OFF);
  float*    bmax = (float*)(ws + BMAX_OFF);
  float*    gmax = (float*)(ws + GMAX_OFF);

  // allow >64KB dynamic LDS (CDNA5 WGP has 320KB)
  (void)hipFuncSetAttribute((const void*)quad_kernel,
                            hipFuncAttributeMaxDynamicSharedMemorySize,
                            SMEM_BYTES);

  cvt_x_kernel<<<1024, 256, 0, stream>>>(X, Xb, N_PTS * N_NODES);
  cvt_ht_kernel<<<dim3(N_NODES / 32, N_NODES / 32, N_CMP), dim3(32, 8), 0,
                  stream>>>(H, Hbt);
  quad_kernel<<<dim3(N_PTS / PTILE, N_CMP), THREADS, SMEM_BYTES, stream>>>(
      Xb, Hbt, kv, en);
  max1_kernel<<<256, 256, 0, stream>>>(en, bmax, N_PTS * N_CMP);
  max2_kernel<<<1, 256, 0, stream>>>(bmax, gmax, 256);
  finalize_kernel<<<2048, 256, 0, stream>>>(en, gmax, out, N_PTS * N_CMP);
}
